// VPGNet_78271484003217
// MI455X (gfx1250) — compile-verified, matrix operand order checked by reading, not run on hardware
//
#include <hip/hip_runtime.h>

// ---------------------------------------------------------------------------
// CDNA5 (gfx1250) VPGNet forward.
// Heavy convs -> implicit GEMM with v_wmma_f32_16x16x32_bf16 (wave32, fp32
// accumulate). K loop restructured as (kh,kw) x ci-chunks: zero div/mod in
// the hot loop, compile-time kernel geometry, hoisted padding checks.
// K-heavy batch-1 layers (N=300) use NSUB=10 tiles so the big weight
// matrices stream from HBM only twice (weight-bound roofline ~23 TB/s).
// ---------------------------------------------------------------------------

typedef __attribute__((ext_vector_type(16))) __bf16 v16bf;
typedef __attribute__((ext_vector_type(8)))  float  v8f;

// ---------------------------------------------------------------------------
// Spatial conv (NCHW batch 1, OIHW weights) as implicit GEMM:
//   M = Cout, N = Hout*Wout, K = Cin*KH*KW, iterated (kh,kw,ci32).
// One wave per block; block computes 16(M) x 16*NSUB(N) of output.
// CHK=true handles Cin not a multiple of 32 (conv1 only).
// ---------------------------------------------------------------------------
template<int KH, int KW, int STRIDE, int PAD, int NSUB, bool CHK>
__global__ void conv_wmma_sp(const float* __restrict__ X,
                             const float* __restrict__ Wt,
                             const float* __restrict__ Bs,
                             float* __restrict__ Y,
                             int Cin, int Hin, int Win,
                             int Cout, int Hout, int Wout, int doRelu)
{
    constexpr int KHW = KH * KW;
    const int lane   = threadIdx.x;        // 0..31
    const int M      = Cout;
    const int N      = Hout * Wout;
    const int HinWin = Hin * Win;

    const int n0 = blockIdx.x * (16 * NSUB);
    const int m0 = blockIdx.y * 16;
    const int lm = lane & 15;
    const int hl = lane >> 4;
    const int kb = hl * 8;

    // Clamp m so weight loads stay in-bounds; invalid rows discarded at store.
    const int m  = m0 + lm;
    const int mc = (m < M) ? m : (M - 1);
    const float* wrow = Wt + (size_t)mc * (size_t)Cin * KHW;

    // Hoisted spatial decode per N sub-tile (clamped; discarded at store).
    int ohv[NSUB], owv[NSUB];
#pragma unroll
    for (int s = 0; s < NSUB; ++s) {
        int n = n0 + s * 16 + lm;
        if (n >= N) n = N - 1;
        ohv[s] = n / Wout;
        owv[s] = n - ohv[s] * Wout;
    }

    // Per-element ci pattern within a 32-wide K chunk, and hoisted offsets.
    int cpat[16], aoff[16], boff[16];
#pragma unroll
    for (int e = 0; e < 16; ++e) {
        cpat[e] = kb + ((e < 8) ? e : (e + 8));   // {kb+0..7, kb+16..23}
        aoff[e] = cpat[e] * KHW;                  // weight stride per ci (const KHW)
        boff[e] = cpat[e] * HinWin;               // activation stride per ci
    }

    v8f acc[NSUB];
#pragma unroll
    for (int s = 0; s < NSUB; ++s) acc[s] = {};

    for (int kh = 0; kh < KH; ++kh) {
        for (int kw = 0; kw < KW; ++kw) {
            const float* wp = wrow + kh * KW + kw;
            for (int c0 = 0; c0 < Cin; c0 += 32) {
                // ---- A fragment: 16x32 bf16 weights (lane = row m) ----
                v16bf a;
#pragma unroll
                for (int e = 0; e < 16; ++e) {
                    float v;
                    if (CHK) {
                        int ci = c0 + cpat[e];
                        v = (ci < Cin) ? wp[(size_t)ci * KHW] : 0.0f;
                    } else {
                        v = wp[(size_t)c0 * KHW + aoff[e]];
                    }
                    a[e] = (__bf16)v;
                }
#pragma unroll
                for (int s = 0; s < NSUB; ++s) {
                    // ---- B fragment: 32x16 bf16 activations (lane = col n) ----
                    int ih = ohv[s] * STRIDE - PAD + kh;
                    int iw = owv[s] * STRIDE - PAD + kw;
                    bool inb = (ih >= 0) && (ih < Hin) && (iw >= 0) && (iw < Win);
                    v16bf b;
#pragma unroll
                    for (int e = 0; e < 16; ++e) {
                        float v = 0.0f;
                        if (inb) {
                            if (CHK) {
                                int ci  = c0 + cpat[e];
                                int cic = (ci < Cin) ? ci : (Cin - 1);  // a[e]==0 there
                                v = X[(size_t)cic * HinWin + ih * Win + iw];
                            } else {
                                v = X[(size_t)c0 * HinWin + ih * Win + iw + boff[e]];
                            }
                        }
                        b[e] = (__bf16)v;
                    }
                    acc[s] = __builtin_amdgcn_wmma_f32_16x16x32_bf16(
                        false, a, false, b, (short)0, acc[s], false, false);
                }
            }
        }
    }

    // ---- store D (+bias, optional ReLU) ----
#pragma unroll
    for (int s = 0; s < NSUB; ++s) {
        int n = n0 + s * 16 + lm;
        if (n >= N) continue;
#pragma unroll
        for (int v = 0; v < 8; ++v) {
            int mm = m0 + v + hl * 8;
            if (mm < M) {
                float val = acc[s][v] + Bs[mm];
                if (doRelu) val = fmaxf(val, 0.0f);
                Y[(size_t)mm * N + n] = val;
            }
        }
    }
}

// ---------------------------------------------------------------------------
// Plain GEMM (1x1 convs / heads): Y[M,N] = W[M,K] * X[K,N] + bias.
// A loads are two contiguous 8-float runs -> mergeable into b128 loads.
// K must be a multiple of 32 (heads: K = 4096).
// ---------------------------------------------------------------------------
template<int NSUB>
__global__ void gemm_wmma(const float* __restrict__ X,
                          const float* __restrict__ Wt,
                          const float* __restrict__ Bs,
                          float* __restrict__ Y,
                          int M, int N, int K, int doRelu)
{
    const int lane = threadIdx.x;
    const int n0 = blockIdx.x * (16 * NSUB);
    const int m0 = blockIdx.y * 16;
    const int lm = lane & 15;
    const int hl = lane >> 4;
    const int kb = hl * 8;

    const int m  = m0 + lm;
    const int mc = (m < M) ? m : (M - 1);
    const float* wp = Wt + (size_t)mc * K;

    int colc[NSUB];
#pragma unroll
    for (int s = 0; s < NSUB; ++s) {
        int n = n0 + s * 16 + lm;
        colc[s] = (n < N) ? n : (N - 1);
    }

    int boff[16];
#pragma unroll
    for (int e = 0; e < 16; ++e)
        boff[e] = (kb + ((e < 8) ? e : (e + 8))) * N;

    v8f acc[NSUB];
#pragma unroll
    for (int s = 0; s < NSUB; ++s) acc[s] = {};

    for (int k0 = 0; k0 < K; k0 += 32) {
        const float* wkb = wp + k0 + kb;     // contiguous runs [0..7], [16..23]
        v16bf a;
#pragma unroll
        for (int e = 0; e < 16; ++e)
            a[e] = (__bf16)wkb[(e < 8) ? e : (e + 8)];
        if (k0 + 32 < K)
            __builtin_prefetch(wkb + 32, 0, 0);   // next weight k-tile
#pragma unroll
        for (int s = 0; s < NSUB; ++s) {
            const float* xk = X + (size_t)k0 * N + colc[s];
            v16bf b;
#pragma unroll
            for (int e = 0; e < 16; ++e)
                b[e] = (__bf16)xk[boff[e]];
            acc[s] = __builtin_amdgcn_wmma_f32_16x16x32_bf16(
                false, a, false, b, (short)0, acc[s], false, false);
        }
    }

#pragma unroll
    for (int s = 0; s < NSUB; ++s) {
        int n = n0 + s * 16 + lm;
        if (n >= N) continue;
#pragma unroll
        for (int v = 0; v < 8; ++v) {
            int mm = m0 + v + hl * 8;
            if (mm < M) {
                float val = acc[s][v] + Bs[mm];
                if (doRelu) val = fmaxf(val, 0.0f);
                Y[(size_t)mm * N + n] = val;
            }
        }
    }
}

// ---- LRN across channels: x / (k + (alpha/size) * sum_win(x^2))^0.75 ----
__global__ void lrn_k(const float* __restrict__ X, float* __restrict__ Y,
                      int C, int HW, float kconst, int total)
{
    int i = blockIdx.x * blockDim.x + threadIdx.x;
    if (i >= total) return;
    int c = i / HW;
    int p = i - c * HW;
    float s = 0.0f;
    for (int dc = -2; dc <= 2; ++dc) {
        int cc = c + dc;
        if (cc >= 0 && cc < C) {
            float v = X[(size_t)cc * HW + p];
            s += v * v;
        }
    }
    Y[i] = X[i] / powf(kconst + 1e-4f * s, 0.75f);  // alpha/size = 0.0005/5
}

// ---- 3x3 stride-2 max pool (VALID) ----
__global__ void maxpool3s2_k(const float* __restrict__ X, float* __restrict__ Y,
                             int C, int Hin, int Win, int Hout, int Wout)
{
    int total = C * Hout * Wout;
    int i = blockIdx.x * blockDim.x + threadIdx.x;
    if (i >= total) return;
    int c  = i / (Hout * Wout);
    int r  = i - c * Hout * Wout;
    int oh = r / Wout;
    int ow = r - oh * Wout;
    float m = -__builtin_inff();
    for (int kh = 0; kh < 3; ++kh)
        for (int kw = 0; kw < 3; ++kw) {
            int ih = oh * 2 + kh;
            int iw = ow * 2 + kw;
            if (ih < Hin && iw < Win)
                m = fmaxf(m, X[(size_t)c * Hin * Win + ih * Win + iw]);
        }
    Y[i] = m;
}

// ---- nearest upsample: src = floor(dst * in/out) ----
__global__ void nearest_k(const float* __restrict__ X, float* __restrict__ Y,
                          int C, int Hin, int Win, int Hout, int Wout)
{
    int total = C * Hout * Wout;
    int i = blockIdx.x * blockDim.x + threadIdx.x;
    if (i >= total) return;
    int c  = i / (Hout * Wout);
    int r  = i - c * Hout * Wout;
    int oh = r / Wout;
    int ow = r - oh * Wout;
    int hi = (oh * Hin) / Hout;
    int wi = (ow * Win) / Wout;
    Y[i] = X[(size_t)c * Hin * Win + hi * Win + wi];
}

// ---- depth-to-space (t x t blocks, row-major) ----
__global__ void d2s_k(const float* __restrict__ X, float* __restrict__ Y,
                      int od, int t, int HS, int WS)
{
    int OH = HS * t, OW = WS * t;
    int total = od * OH * OW;
    int i = blockIdx.x * blockDim.x + threadIdx.x;
    if (i >= total) return;
    int o = i / (OH * OW);
    int r = i - o * OH * OW;
    int y = r / OW;
    int x = r - y * OW;
    int c = o * t * t + (y % t) * t + (x % t);
    Y[i] = X[(size_t)c * HS * WS + (y / t) * WS + (x / t)];
}

// ---------------------------------------------------------------------------

extern "C" void kernel_launch(void* const* d_in, const int* in_sizes, int n_in,
                              void* d_out, int out_size, void* d_ws, size_t ws_size,
                              hipStream_t stream)
{
    (void)in_sizes; (void)n_in; (void)out_size; (void)ws_size;

    const float* x   = (const float*)d_in[0];
    const float* w1  = (const float*)d_in[1];  const float* b1  = (const float*)d_in[2];
    const float* w2  = (const float*)d_in[3];  const float* b2  = (const float*)d_in[4];
    const float* w3  = (const float*)d_in[5];  const float* b3  = (const float*)d_in[6];
    const float* w4  = (const float*)d_in[7];  const float* b4  = (const float*)d_in[8];
    const float* w5  = (const float*)d_in[9];  const float* b5  = (const float*)d_in[10];
    const float* w6  = (const float*)d_in[11]; const float* b6  = (const float*)d_in[12];
    const float* w7[4] = { (const float*)d_in[13], (const float*)d_in[17],
                           (const float*)d_in[21], (const float*)d_in[25] };
    const float* b7[4] = { (const float*)d_in[14], (const float*)d_in[18],
                           (const float*)d_in[22], (const float*)d_in[26] };
    const float* w8[4] = { (const float*)d_in[15], (const float*)d_in[19],
                           (const float*)d_in[23], (const float*)d_in[27] };
    const float* b8[4] = { (const float*)d_in[16], (const float*)d_in[20],
                           (const float*)d_in[24], (const float*)d_in[28] };

    float* out  = (float*)d_out;
    float* bufA = (float*)d_ws;              // 1,789,824 floats
    float* bufB = bufA + 1789824;            // 1,789,824 floats  (~14.3 MB total)

    auto grid1d = [](int total) { return dim3((total + 255) / 256); };

    // conv1: [3,480,640] -> [96,118,158], 11x11 s4 p0, ReLU  (Cin=3 -> CHK)
    { dim3 g((18644 + 31) / 32, (96 + 15) / 16);
      conv_wmma_sp<11, 11, 4, 0, 2, true><<<g, 32, 0, stream>>>(
          x, w1, b1, bufA, 3, 480, 640, 96, 118, 158, 1); }
    // LRN k=2
    { int tot = 96 * 118 * 158;
      lrn_k<<<grid1d(tot), 256, 0, stream>>>(bufA, bufB, 96, 118 * 158, 2.0f, tot); }
    // maxpool -> [96,58,78]
    maxpool3s2_k<<<grid1d(96 * 58 * 78), 256, 0, stream>>>(bufB, bufA, 96, 118, 158, 58, 78);
    // nearest -> [96,59,79]
    nearest_k<<<grid1d(96 * 59 * 79), 256, 0, stream>>>(bufA, bufB, 96, 58, 78, 59, 79);
    // conv2: -> [256,59,79], 5x5 p2, ReLU
    { dim3 g((4661 + 31) / 32, (256 + 15) / 16);
      conv_wmma_sp<5, 5, 1, 2, 2, false><<<g, 32, 0, stream>>>(
          bufB, w2, b2, bufA, 96, 59, 79, 256, 59, 79, 1); }
    // LRN k=8
    { int tot = 256 * 59 * 79;
      lrn_k<<<grid1d(tot), 256, 0, stream>>>(bufA, bufB, 256, 59 * 79, 8.0f, tot); }
    // maxpool -> [256,29,39]
    maxpool3s2_k<<<grid1d(256 * 29 * 39), 256, 0, stream>>>(bufB, bufA, 256, 59, 79, 29, 39);
    // conv3/4/5: 3x3 p1, ReLU, [384,29,39]
    { dim3 g((1131 + 31) / 32, (384 + 15) / 16);
      conv_wmma_sp<3, 3, 1, 1, 2, false><<<g, 32, 0, stream>>>(
          bufA, w3, b3, bufB, 256, 29, 39, 384, 29, 39, 1);
      conv_wmma_sp<3, 3, 1, 1, 2, false><<<g, 32, 0, stream>>>(
          bufB, w4, b4, bufA, 384, 29, 39, 384, 29, 39, 1);
      conv_wmma_sp<3, 3, 1, 1, 2, false><<<g, 32, 0, stream>>>(
          bufA, w5, b5, bufB, 384, 29, 39, 384, 29, 39, 1); }
    // maxpool -> [384,14,19]
    maxpool3s2_k<<<grid1d(384 * 14 * 19), 256, 0, stream>>>(bufB, bufA, 384, 29, 39, 14, 19);
    // conv6 ("shared"): 6x6 p3 -> [4096,15,20], ReLU, straight into d_out[0].
    // N=300, K-heavy (226MB weights): NSUB=10 -> weights streamed only 2x.
    { dim3 g((300 + 159) / 160, (4096 + 15) / 16);
      conv_wmma_sp<6, 6, 1, 3, 10, false><<<g, 32, 0, stream>>>(
          bufA, w6, b6, out, 384, 14, 19, 4096, 15, 20, 1); }

    // heads: 1x1 conv (ReLU) -> 1x1 conv -> depth-to-space into d_out segments
    const int    c8[4]     = { 256, 128, 1024, 320 };
    const int    tt[4]     = { 8, 8, 4, 8 };
    const size_t outOff[4] = { 1228800,                            // grid  [4,120,160]
                               1228800 + 76800,                    // obj   [2,120,160]
                               1228800 + 76800 + 38400,            // multi [64,60,80]
                               1228800 + 76800 + 38400 + 307200 }; // vpp   [5,120,160]

    for (int h = 0; h < 4; ++h) {
        { dim3 g((300 + 159) / 160, (4096 + 15) / 16);
          gemm_wmma<10><<<g, 32, 0, stream>>>(out,  w7[h], b7[h], bufB,
                                              4096, 300, 4096, 1); }
        { dim3 g((300 + 159) / 160, (c8[h] + 15) / 16);
          gemm_wmma<10><<<g, 32, 0, stream>>>(bufB, w8[h], b8[h], bufA,
                                              c8[h], 300, 4096, 0); }
        int od  = c8[h] / (tt[h] * tt[h]);
        int tot = od * 15 * tt[h] * 20 * tt[h];
        d2s_k<<<grid1d(tot), 256, 0, stream>>>(bufA, out + outOff[h], od, tt[h], 15, 20);
    }
}
